// QLinearPacked_47588237640427
// MI455X (gfx1250) — matrix operand order repaired and checked by reading, compiled, not run on hardware
//
#include <hip/hip_runtime.h>

typedef __attribute__((ext_vector_type(16))) _Float16 v16h;
typedef __attribute__((ext_vector_type(8)))  _Float16 v8h;
typedef __attribute__((ext_vector_type(8)))  float    v8f;
typedef __attribute__((ext_vector_type(4)))  int      v4i;

#define M_DIM   32
#define K_DIM   4096
#define N_DIM   11008
#define KSPLIT  4
#define KCHUNK  (K_DIM / KSPLIT)   // 1024 per wave
#define EXPOFF  133                // 127 (E8M0 bias) + 6 (Q1.6 mantissa)

// ---------------- x: f32 -> f16 staging (runs once per call, L2-resident) ---
__global__ void x_cvt_f16(const float* __restrict__ x, _Float16* __restrict__ x16, int n) {
    int i = blockIdx.x * blockDim.x + threadIdx.x;
    if (i < n) x16[i] = (_Float16)x[i];
}

// sign-extend int8 stored in low byte of int32, scale, convert to f16
__device__ __forceinline__ void dq4(v4i q, float s, v16h& b, int base) {
    b[base + 0] = (_Float16)((float)((q[0] << 24) >> 24) * s);
    b[base + 1] = (_Float16)((float)((q[1] << 24) >> 24) * s);
    b[base + 2] = (_Float16)((float)((q[2] << 24) >> 24) * s);
    b[base + 3] = (_Float16)((float)((q[3] << 24) >> 24) * s);
}

// ---------------- main: MXINT8 dequant + WMMA GEMM, 4-way K-split ----------
__launch_bounds__(128, 1)
__global__ void qlinear_wmma(const _Float16* __restrict__ x16,
                             const int* __restrict__ wq,
                             const int* __restrict__ wsc,
                             const int* __restrict__ bq,
                             const int* __restrict__ bsc,
                             float* __restrict__ out)
{
    __shared__ float red[KSPLIT][2][8][32];           // 8 KB partial-sum exchange

    const int lane = threadIdx.x & 31;
    const int wv   = threadIdx.x >> 5;                // K-split index 0..3
    const int n_base = blockIdx.x << 4;               // 16 out-features per block
    const int col  = lane & 15;
    const int hi   = lane >> 4;                       // 0: lanes 0-15, 1: lanes 16-31
    const int aoff = hi << 3;                         // unused; A loads use boff (16h) too
    const int boff = hi << 4;                         // B-fragment K sub-offset
    const int n    = n_base + col;

    const _Float16* xrow0 = x16 + (size_t)col * K_DIM;          // M rows 0..15
    const _Float16* xrow1 = x16 + (size_t)(col + 16) * K_DIM;   // M rows 16..31
    const int* wrow = wq  + (size_t)n * K_DIM;
    const int* srow = wsc + (size_t)n * (K_DIM / 16);

    v8f acc0 = {};  // D tile for M 0..15
    v8f acc1 = {};  // D tile for M 16..31

    int k0 = wv * KCHUNK;
#pragma unroll 2
    for (int it = 0; it < KCHUNK / 32; ++it, k0 += 32) {
        const int kb = k0 + boff;                     // this lane's 16-wide MX group

        // stream-ahead prefetch on the weight stream (global_prefetch_b8)
        __builtin_prefetch(wrow + kb + 512, 0, 1);

        // ---- A fragments: plain b128 loads from row-major f16 x ----
        v8h a0lo = *(const v8h*)(xrow0 + k0 + boff);
        v8h a0hi = *(const v8h*)(xrow0 + k0 + 8 + boff);
        v8h a1lo = *(const v8h*)(xrow1 + k0 + boff);
        v8h a1hi = *(const v8h*)(xrow1 + k0 + 8 + boff);

        // ---- B fragment: 16 consecutive int32 mantissas, one shared exponent ----
        v4i q0 = *(const v4i*)(wrow + kb);
        v4i q1 = *(const v4i*)(wrow + kb + 4);
        v4i q2 = *(const v4i*)(wrow + kb + 8);
        v4i q3 = *(const v4i*)(wrow + kb + 12);
        const float s = __builtin_ldexpf(1.0f, srow[kb >> 4] - EXPOFF);

        v16h b;
        dq4(q0, s, b, 0);  dq4(q1, s, b, 4);
        dq4(q2, s, b, 8);  dq4(q3, s, b, 12);

        v16h a0, a1;
#pragma unroll
        for (int j = 0; j < 8; ++j) {
            a0[j] = a0lo[j];  a0[j + 8] = a0hi[j];
            a1[j] = a1lo[j];  a1[j + 8] = a1hi[j];
        }

        acc0 = __builtin_amdgcn_wmma_f32_16x16x32_f16(false, a0, false, b,
                                                      (short)0, acc0, false, false);
        acc1 = __builtin_amdgcn_wmma_f32_16x16x32_f16(false, a1, false, b,
                                                      (short)0, acc1, false, false);
    }

    // ---- cross-wave K reduction through LDS ----
#pragma unroll
    for (int r = 0; r < 8; ++r) {
        red[wv][0][r][lane] = acc0[r];
        red[wv][1][r][lane] = acc1[r];
    }
    __syncthreads();

    if (wv == 0) {
        // dequantize bias for this lane's out-feature
        const float bias = (float)((bq[n] << 24) >> 24) *
                           __builtin_ldexpf(1.0f, bsc[n >> 4] - EXPOFF);
#pragma unroll
        for (int r = 0; r < 8; ++r) {
            float s0 = red[0][0][r][lane] + red[1][0][r][lane] +
                       red[2][0][r][lane] + red[3][0][r][lane] + bias;
            float s1 = red[0][1][r][lane] + red[1][1][r][lane] +
                       red[2][1][r][lane] + red[3][1][r][lane] + bias;
            const int m = hi ? (r + 8) : r;           // C/D layout: lanes 16-31 -> M+8
            out[(size_t)m        * N_DIM + n] = s0;
            out[(size_t)(m + 16) * N_DIM + n] = s1;
        }
    }
}

extern "C" void kernel_launch(void* const* d_in, const int* in_sizes, int n_in,
                              void* d_out, int out_size, void* d_ws, size_t ws_size,
                              hipStream_t stream) {
    const float* x   = (const float*)d_in[0];
    const int*   wq  = (const int*)d_in[1];
    const int*   wsc = (const int*)d_in[2];
    const int*   bq  = (const int*)d_in[3];
    const int*   bsc = (const int*)d_in[4];
    float* out = (float*)d_out;

    _Float16* x16 = (_Float16*)d_ws;                  // needs 32*4096*2 = 256 KB
    const int nx = M_DIM * K_DIM;
    x_cvt_f16<<<(nx + 255) / 256, 256, 0, stream>>>(x, x16, nx);

    qlinear_wmma<<<N_DIM / 16, 128, 0, stream>>>(x16, wq, wsc, bq, bsc, out);
}
